// MPNNBlock_8589934592463
// MI455X (gfx1250) — compile-verified, matrix operand order checked by reading, not run on hardware
//
#include <hip/hip_runtime.h>
#include <cstddef>

// ---------------------------------------------------------------------------
// MPNN forward for MI455X (gfx1250, wave32, WMMA).
//
// Roofline: materializing We [E,64,64] f32 costs ~2 GB HBM traffic (~86us at
// 23.3 TB/s). Instead we FUSE: msgs = sum_k rf[e,k] * (h_src @ W2_k),
// recomputing We implicitly per layer with v_wmma_f32_16x16x32_bf16. The bf16
// pre-swizzled en_W2 (1 MB) stays L2-resident (192 MB L2). Inner-loop A-scaling
// uses v_pk_mul_bf16 (inline asm, VOP3P, non-volatile / schedulable); the
// k-loop is unrolled 2x with a hand software-pipelined ds_load_b64 feeding the
// NEXT iteration so the LDS latency is hidden behind a full WMMA iteration.
// ---------------------------------------------------------------------------

typedef __attribute__((ext_vector_type(16))) __bf16 v16bf;
typedef __attribute__((ext_vector_type(8)))  float  v8f;
typedef __attribute__((ext_vector_type(4)))  unsigned int v4u;
typedef __attribute__((ext_vector_type(2)))  unsigned int v2u;

union BF16x16 {
    v16bf v;
    unsigned short us[16];
    unsigned int   u32[8];
    v4u q[2];
};

__device__ __forceinline__ unsigned short f2bf(float x) {
    unsigned u = __float_as_uint(x);
    u += 0x7fffu + ((u >> 16) & 1u);   // round to nearest even
    return (unsigned short)(u >> 16);
}

// packed bf16 multiply: d.{lo,hi} = a.{lo,hi} * b.{lo,hi}  (VOP3P v_pk_mul_bf16)
// NOT volatile: pure computation, keep it schedulable for software pipelining.
__device__ __forceinline__ unsigned pk_mul_bf16(unsigned a, unsigned b) {
    unsigned d;
    asm("v_pk_mul_bf16 %0, %1, %2" : "=v"(d) : "v"(a), "v"(b));
    return d;
}

__device__ __forceinline__ void loadfrag(BF16x16& d, const unsigned short* p) {
    const v4u* q = (const v4u*)p;      // 32-byte aligned fragment
    d.q[0] = q[0];
    d.q[1] = q[1];
}

__device__ __forceinline__ v8f wmma_bf16(const BF16x16& a, const BF16x16& b, v8f c) {
    return __builtin_amdgcn_wmma_f32_16x16x32_bf16(
        false, a.v, false, b.v, (short)0, c, false, false);
}

__device__ __forceinline__ float sigm(float x) { return 1.f / (1.f + __expf(-x)); }

// K-offset inside a 32-wide K chunk for 16-bit A/B operand fragments:
// lane half (lane>>4) offsets K by 8; element t maps t<8 -> t, t>=8 -> t+8.
__device__ __forceinline__ int kmap(int t, int half) {
    return ((t < 8) ? t : t + 8) + half * 8;
}

// monotone float -> uint mapping for atomic segment-max
__device__ __forceinline__ unsigned ordf(float f) {
    unsigned u = __float_as_uint(f);
    return (u & 0x80000000u) ? ~u : (u | 0x80000000u);
}
__device__ __forceinline__ float unordf(unsigned u) {
    unsigned v = (u & 0x80000000u) ? (u & 0x7fffffffu) : ~u;
    return __uint_as_float(v);
}

#define ZERO8 v8f{0.f,0.f,0.f,0.f,0.f,0.f,0.f,0.f}
#define RFS 132   // rf LDS row stride in dwords: 8B/16B-aligned rows, banks (4r+k)%64

// ---------------------------------------------------------------------------
// Small helper kernels
// ---------------------------------------------------------------------------
__global__ void k_zero_f(float* p, int n) {
    int i = blockIdx.x * blockDim.x + threadIdx.x;
    if (i < n) p[i] = 0.f;
}
__global__ void k_zero_u(unsigned* p, int n) {
    int i = blockIdx.x * blockDim.x + threadIdx.x;
    if (i < n) p[i] = 0u;
}

__global__ void k_init_h(const float* __restrict__ x, float* __restrict__ h, int N) {
    int i = blockIdx.x * blockDim.x + threadIdx.x;
    if (i >= N * 64) return;
    int n = i >> 6, j = i & 63;
    h[i] = (j < 14) ? x[n * 14 + j] : 0.f;
}

__global__ void k_deg(const int* __restrict__ etgt, int* deg, int E) {
    int i = blockIdx.x * blockDim.x + threadIdx.x;
    if (i < E) atomicAdd(&deg[etgt[i]], 1);
}
__global__ void k_invdeg(const int* __restrict__ deg, float* inv, int N) {
    int i = blockIdx.x * blockDim.x + threadIdx.x;
    if (i < N) inv[i] = 1.f / fmaxf((float)deg[i], 1.f);
}

// rf = relu(edge_attr @ en_W1 + en_b1)   [E,128] fp32
__global__ void k_edge_rf(const float* __restrict__ ea, const float* __restrict__ W1,
                          const float* __restrict__ b1, float* __restrict__ rf, int E) {
    int i = blockIdx.x * blockDim.x + threadIdx.x;
    if (i >= E * 128) return;
    int e = i >> 7, c = i & 127;
    float s = b1[c];
#pragma unroll
    for (int t = 0; t < 4; ++t) s += ea[e * 4 + t] * W1[t * 128 + c];
    rf[i] = fmaxf(s, 0.f);
}

// Pre-swizzle a row-major [nk x 64 x ncols] weight into wave32 16-bit
// B-operand fragments: dst[((k*2+kk)*ct_count + ct)*512 + lane*16 + t]
__global__ void k_pack_frag(const float* __restrict__ src, unsigned short* __restrict__ dst,
                            int ncols, int ct_count, int total) {
    int idx = blockIdx.x * blockDim.x + threadIdx.x;
    if (idx >= total) return;
    int per_k = 2 * ct_count * 512;
    int k   = idx / per_k;
    int rem = idx % per_k;
    int kk  = rem / (ct_count * 512);
    int r2  = rem % (ct_count * 512);
    int ct  = r2 / 512;
    int li  = r2 % 512;
    int l = li >> 4, t = li & 15;
    int halfl = l >> 4, n = l & 15;
    int i = kk * 32 + kmap(t, halfl);
    dst[idx] = f2bf(src[(size_t)k * 64 * ncols + (size_t)i * ncols + ct * 16 + n]);
}

// ---------------------------------------------------------------------------
// Fused edge-network + message + scatter kernel (WMMA heavy).
// One wave per 16-edge tile; msgs(16x64) = Hsrc @ B2 + sum_k rf_k*Hsrc @ W2_k
// ---------------------------------------------------------------------------
__global__ __launch_bounds__(32)
void k_edge_msg(const float* __restrict__ rf,             // [E,128]
                const float* __restrict__ h,              // [N,64]
                const unsigned short* __restrict__ w2frag,// [128][2][4][512]
                const unsigned short* __restrict__ b2frag,// [2][4][512]
                const int* __restrict__ esrc, const int* __restrict__ etgt,
                float* __restrict__ agg, int E) {
    __shared__ unsigned rf_lds[16 * RFS];   // broadcast-packed bf16 pairs {bf,bf}
    const int lane = threadIdx.x;
    const int half = lane >> 4;
    const int col  = lane & 15;
    const int ebase = blockIdx.x * 16;

    // stage rf tile, pre-converted to broadcast-packed bf16
    for (int idx = lane; idx < 16 * 128; idx += 32) {
        int e = idx >> 7, k = idx & 127;
        int eg = ebase + e; if (eg >= E) eg = E - 1;
        unsigned b = (unsigned)f2bf(rf[eg * 128 + k]);
        rf_lds[e * RFS + k] = b | (b << 16);
    }
    __syncthreads();

    // gather h_src row for this lane and pack to bf16 pairs (once per tile)
    int eg = ebase + col; if (eg >= E) eg = E - 1;
    const int s = esrc[eg];
    unsigned hb0[8], hb1[8];
#pragma unroll
    for (int j = 0; j < 8; ++j) {
        int i0 = kmap(2 * j, half), i1 = kmap(2 * j + 1, half);
        hb0[j] = (unsigned)f2bf(h[s * 64 + i0]) | ((unsigned)f2bf(h[s * 64 + i1]) << 16);
        hb1[j] = (unsigned)f2bf(h[s * 64 + 32 + i0]) |
                 ((unsigned)f2bf(h[s * 64 + 32 + i1]) << 16);
    }

    // bias pass: msgs += Hsrc @ B2
    BF16x16 a0, a1;
#pragma unroll
    for (int j = 0; j < 8; ++j) { a0.u32[j] = hb0[j]; a1.u32[j] = hb1[j]; }
    v8f acc[4];
#pragma unroll
    for (int nt = 0; nt < 4; ++nt) {
        BF16x16 b0, b1;
        loadfrag(b0, b2frag + (0 * 4 + nt) * 512 + lane * 16);
        loadfrag(b1, b2frag + (1 * 4 + nt) * 512 + lane * 16);
        v8f z = ZERO8;
        acc[nt] = wmma_bf16(a0, b0, z);
        acc[nt] = wmma_bf16(a1, b1, acc[nt]);
    }

    // K loop over the 128 hidden features of the edge MLP, 2 k's per iteration.
    // Software pipeline: the ds_load_b64 for iteration i+1 is issued before the
    // pk_mul/WMMA work of iteration i, so s_wait_dscnt retires a full iteration
    // after its load and the LDS latency is hidden.
    const unsigned short* bbase = w2frag + lane * 16;
    const unsigned* rfrow = rf_lds + col * RFS;
    v2u rr2 = *(const v2u*)(rfrow);            // prologue: pair for k=0,1
    for (int k = 0; k < 128; k += 2) {
        v2u cur = rr2;
        if (k + 2 < 128)
            rr2 = *(const v2u*)(rfrow + k + 2);   // next pair (8B-aligned)
        if (k + 8 < 128)   // keep the 1 MB B stream ahead (global_prefetch_b8)
            __builtin_prefetch(bbase + (size_t)(k + 8) * 8 * 512, 0, 1);
#pragma unroll
        for (int u = 0; u < 2; ++u) {
            unsigned rr = cur[u];
            BF16x16 sa0, sa1;
#pragma unroll
            for (int j = 0; j < 8; ++j) {
                sa0.u32[j] = pk_mul_bf16(hb0[j], rr);
                sa1.u32[j] = pk_mul_bf16(hb1[j], rr);
            }
            const unsigned short* bp = bbase + (size_t)(k + u) * 8 * 512;
#pragma unroll
            for (int nt = 0; nt < 4; ++nt) {
                BF16x16 b0, b1;
                loadfrag(b0, bp + (0 * 4 + nt) * 512);
                loadfrag(b1, bp + (1 * 4 + nt) * 512);
                acc[nt] = wmma_bf16(sa0, b0, acc[nt]);
                acc[nt] = wmma_bf16(sa1, b1, acc[nt]);
            }
        }
    }

    // scatter-add into agg[tgt] (C layout: row = r + half*8, col = nt*16 + col)
#pragma unroll
    for (int r = 0; r < 8; ++r) {
        int erow = ebase + r + half * 8;
        if (erow >= E) continue;
        int tg = etgt[erow];
#pragma unroll
        for (int nt = 0; nt < 4; ++nt)
            atomicAdd(&agg[tg * 64 + nt * 16 + col], acc[nt][r]);
    }
}

// ---------------------------------------------------------------------------
// Node update: m = agg*inv_deg + h@root_W + conv_b ; GRU(h, m)  (WMMA)
// One wave per 16-node tile.
// ---------------------------------------------------------------------------
__global__ __launch_bounds__(32)
void k_node_gru(const float* __restrict__ h, const float* __restrict__ agg,
                const float* __restrict__ inv_deg,
                const unsigned short* __restrict__ rootfrag,  // [2][4][512]
                const unsigned short* __restrict__ wihfrag,   // [2][12][512]
                const unsigned short* __restrict__ whhfrag,   // [2][12][512]
                const float* __restrict__ conv_b,
                const float* __restrict__ b_ih, const float* __restrict__ b_hh,
                float* __restrict__ h_out, float* __restrict__ out_nodes, int N) {
    __shared__ float hl[16 * 65];
    __shared__ float ml[16 * 65];
    const int lane = threadIdx.x;
    const int half = lane >> 4;
    const int col  = lane & 15;
    const int nbase = blockIdx.x * 16;

    for (int idx = lane; idx < 16 * 64; idx += 32) {
        int rr = idx >> 6, c = idx & 63;
        int row = nbase + rr; if (row >= N) row = N - 1;
        hl[rr * 65 + c] = h[row * 64 + c];
    }
    __syncthreads();

    BF16x16 ah0, ah1;
#pragma unroll
    for (int t = 0; t < 16; ++t) {
        int i = kmap(t, half);
        ah0.us[t] = f2bf(hl[col * 65 + i]);
        ah1.us[t] = f2bf(hl[col * 65 + 32 + i]);
    }

    float idg[8];
#pragma unroll
    for (int r = 0; r < 8; ++r) {
        int row = nbase + r + half * 8; if (row >= N) row = N - 1;
        idg[r] = inv_deg[row];
    }

    // m = h @ root_W + agg*inv_deg + conv_b, staged through LDS to re-layout
#pragma unroll
    for (int nt = 0; nt < 4; ++nt) {
        BF16x16 b0, b1;
        loadfrag(b0, rootfrag + (0 * 4 + nt) * 512 + lane * 16);
        loadfrag(b1, rootfrag + (1 * 4 + nt) * 512 + lane * 16);
        v8f z = ZERO8;
        v8f c0 = wmma_bf16(ah0, b0, z);
        c0 = wmma_bf16(ah1, b1, c0);
        float cb = conv_b[nt * 16 + col];
#pragma unroll
        for (int r = 0; r < 8; ++r) {
            int row = nbase + r + half * 8; if (row >= N) row = N - 1;
            float m = c0[r] + agg[row * 64 + nt * 16 + col] * idg[r] + cb;
            ml[(r + half * 8) * 65 + nt * 16 + col] = m;
        }
    }
    __syncthreads();

    BF16x16 am0, am1;
#pragma unroll
    for (int t = 0; t < 16; ++t) {
        int i = kmap(t, half);
        am0.us[t] = f2bf(ml[col * 65 + i]);
        am1.us[t] = f2bf(ml[col * 65 + 32 + i]);
    }

    // GRU gates: gi = m@W_ih, gh = h@W_hh  (192 cols = 12 col tiles, 4 per gate)
#pragma unroll
    for (int nb = 0; nb < 4; ++nb) {
        v8f gi_[3], gh_[3];
#pragma unroll
        for (int g = 0; g < 3; ++g) {
            int ct = g * 4 + nb;
            BF16x16 b0, b1;
            loadfrag(b0, wihfrag + (0 * 12 + ct) * 512 + lane * 16);
            loadfrag(b1, wihfrag + (1 * 12 + ct) * 512 + lane * 16);
            v8f z = ZERO8;
            v8f gi = wmma_bf16(am0, b0, z);
            gi = wmma_bf16(am1, b1, gi);
            loadfrag(b0, whhfrag + (0 * 12 + ct) * 512 + lane * 16);
            loadfrag(b1, whhfrag + (1 * 12 + ct) * 512 + lane * 16);
            v8f z2 = ZERO8;
            v8f gh = wmma_bf16(ah0, b0, z2);
            gh = wmma_bf16(ah1, b1, gh);
            gi_[g] = gi; gh_[g] = gh;
        }
        int c = nb * 16 + col;
        float bir = b_ih[c], biz = b_ih[64 + c], bin = b_ih[128 + c];
        float bhr = b_hh[c], bhz = b_hh[64 + c], bhn = b_hh[128 + c];
#pragma unroll
        for (int r = 0; r < 8; ++r) {
            float rr = sigm(gi_[0][r] + bir + gh_[0][r] + bhr);
            float zz = sigm(gi_[1][r] + biz + gh_[1][r] + bhz);
            float nn = tanhf(gi_[2][r] + bin + rr * (gh_[2][r] + bhn));
            float hold = hl[(r + half * 8) * 65 + c];
            float hnew = (1.f - zz) * nn + zz * hold;
            int row = nbase + r + half * 8;
            if (row < N) {
                h_out[row * 64 + c] = hnew;
                if (out_nodes) out_nodes[row * 64 + c] = hnew;
            }
        }
    }
}

// ---------------------------------------------------------------------------
// Set2Set + readout (small: B=256)
// ---------------------------------------------------------------------------
__global__ void k_lstm(const float* __restrict__ qs, const float* __restrict__ wih,
                       const float* __restrict__ whh, const float* __restrict__ bih,
                       const float* __restrict__ bhh, const float* __restrict__ hh_in,
                       float* __restrict__ hh_out, float* __restrict__ cc,
                       float* __restrict__ qs_out, int Bg) {
    int idx = blockIdx.x * blockDim.x + threadIdx.x;
    if (idx >= Bg * 64) return;
    int b = idx >> 6, j = idx & 63;
    float g[4];
#pragma unroll
    for (int gg = 0; gg < 4; ++gg) {
        float s = bih[gg * 64 + j] + bhh[gg * 64 + j];
        for (int i = 0; i < 128; ++i) s += qs[b * 128 + i] * wih[i * 256 + gg * 64 + j];
        for (int i = 0; i < 64;  ++i) s += hh_in[b * 64 + i] * whh[i * 256 + gg * 64 + j];
        g[gg] = s;
    }
    float c_new = sigm(g[1]) * cc[idx] + sigm(g[0]) * tanhf(g[2]);
    float h_new = sigm(g[3]) * tanhf(c_new);
    cc[idx] = c_new;
    hh_out[b * 64 + j] = h_new;
    qs_out[b * 128 + j] = h_new;   // q part of next q_star
}

__global__ void k_s2s_init(unsigned* mx, float* denom, float* qs_out, int Bg) {
    int idx = blockIdx.x * blockDim.x + threadIdx.x;
    if (idx >= Bg * 64) return;
    int b = idx >> 6, j = idx & 63;
    qs_out[b * 128 + 64 + j] = 0.f;
    if (j == 0) { mx[b] = 0u; denom[b] = 0.f; }
}

__global__ void k_dotmax(const float* __restrict__ nodes, const float* __restrict__ hh,
                         const int* __restrict__ batch, float* __restrict__ evals,
                         unsigned* mx, int N) {
    int n = blockIdx.x * blockDim.x + threadIdx.x;
    if (n >= N) return;
    int b = batch[n];
    float s = 0.f;
    for (int j = 0; j < 64; ++j) s += nodes[n * 64 + j] * hh[b * 64 + j];
    evals[n] = s;
    atomicMax(&mx[b], ordf(s));
}

__global__ void k_expsum(const float* __restrict__ evals, const unsigned* __restrict__ mx,
                         const int* __restrict__ batch, float* __restrict__ avals,
                         float* denom, int N) {
    int n = blockIdx.x * blockDim.x + threadIdx.x;
    if (n >= N) return;
    int b = batch[n];
    float a = __expf(evals[n] - unordf(mx[b]));
    avals[n] = a;
    atomicAdd(&denom[b], a);
}

__global__ void k_weighted(const float* __restrict__ avals, const float* __restrict__ denom,
                           const int* __restrict__ batch, const float* __restrict__ nodes,
                           float* qs_out, int N) {
    int n = blockIdx.x * blockDim.x + threadIdx.x;
    if (n >= N) return;
    int b = batch[n];
    float w = avals[n] / fmaxf(denom[b], 1e-38f);
    for (int j = 0; j < 64; ++j)
        atomicAdd(&qs_out[b * 128 + 64 + j], w * nodes[n * 64 + j]);
}

__global__ void k_ro1(const float* __restrict__ qs, const float* __restrict__ W1,
                      const float* __restrict__ b1, float* __restrict__ t, int Bg) {
    int idx = blockIdx.x * blockDim.x + threadIdx.x;
    if (idx >= Bg * 64) return;
    int b = idx >> 6, j = idx & 63;
    float s = b1[j];
    for (int i = 0; i < 128; ++i) s += qs[b * 128 + i] * W1[i * 64 + j];
    t[idx] = fmaxf(s, 0.f);
}

__global__ void k_ro2(const float* __restrict__ t, const float* __restrict__ W2,
                      const float* __restrict__ b2, float* __restrict__ out, int Bg) {
    int idx = blockIdx.x * blockDim.x + threadIdx.x;
    if (idx >= Bg * 64) return;
    int b = idx >> 6, j = idx & 63;
    float s = b2[j];
    for (int i = 0; i < 64; ++i) s += t[b * 64 + i] * W2[i * 64 + j];
    out[idx] = s;
}

// ---------------------------------------------------------------------------
// Host orchestration
// ---------------------------------------------------------------------------
extern "C" void kernel_launch(void* const* d_in, const int* in_sizes, int n_in,
                              void* d_out, int out_size, void* d_ws, size_t ws_size,
                              hipStream_t stream) {
    const float* x      = (const float*)d_in[0];
    const int*   eidx   = (const int*)  d_in[1];
    const int*   batch  = (const int*)  d_in[3];
    const float* en_W1  = (const float*)d_in[4];
    const float* en_b1  = (const float*)d_in[5];
    const float* en_W2  = (const float*)d_in[6];
    const float* en_b2  = (const float*)d_in[7];
    const float* root_W = (const float*)d_in[8];
    const float* conv_b = (const float*)d_in[9];
    const float* g_wih  = (const float*)d_in[10];
    const float* g_whh  = (const float*)d_in[11];
    const float* g_bih  = (const float*)d_in[12];
    const float* g_bhh  = (const float*)d_in[13];
    const float* l_wih  = (const float*)d_in[14];
    const float* l_whh  = (const float*)d_in[15];
    const float* l_bih  = (const float*)d_in[16];
    const float* l_bhh  = (const float*)d_in[17];
    const float* ro_W1  = (const float*)d_in[18];
    const float* ro_b1  = (const float*)d_in[19];
    const float* ro_W2  = (const float*)d_in[20];
    const float* ro_b2  = (const float*)d_in[21];

    const int N  = in_sizes[0] / 14;     // 10000
    const int E  = in_sizes[1] / 2;      // 30000
    const int Bg = out_size / 64 - N;    // 256
    const int* esrc = eidx;
    const int* etgt = eidx + E;

    // ---- workspace carve (~25 MB) ----
    char* w = (char*)d_ws;
    auto carve = [&](size_t bytes) -> char* {
        char* p = w;
        w += (bytes + 255) & ~(size_t)255;
        return p;
    };
    float* hA       = (float*)carve((size_t)N * 64 * 4);
    float* hB       = (float*)carve((size_t)N * 64 * 4);
    float* agg      = (float*)carve((size_t)N * 64 * 4);
    float* rf       = (float*)carve((size_t)E * 128 * 4);
    float* invdeg   = (float*)carve((size_t)N * 4);
    int*   deg      = (int*)  carve((size_t)N * 4);
    unsigned short* w2frag   = (unsigned short*)carve((size_t)128 * 8 * 512 * 2);
    unsigned short* b2frag   = (unsigned short*)carve((size_t)8 * 512 * 2);
    unsigned short* rootfrag = (unsigned short*)carve((size_t)8 * 512 * 2);
    unsigned short* wihfrag  = (unsigned short*)carve((size_t)24 * 512 * 2);
    unsigned short* whhfrag  = (unsigned short*)carve((size_t)24 * 512 * 2);
    float* qs0    = (float*)carve((size_t)Bg * 128 * 4);
    float* qs1    = (float*)carve((size_t)Bg * 128 * 4);
    float* hh0    = (float*)carve((size_t)Bg * 64 * 4);
    float* hh1    = (float*)carve((size_t)Bg * 64 * 4);
    float* ccb    = (float*)carve((size_t)Bg * 64 * 4);
    float* evals  = (float*)carve((size_t)N * 4);
    float* avals  = (float*)carve((size_t)N * 4);
    unsigned* mx  = (unsigned*)carve((size_t)Bg * 4);
    float* denom  = (float*)carve((size_t)Bg * 4);
    float* tbuf   = (float*)carve((size_t)Bg * 64 * 4);
    (void)ws_size; (void)n_in;

    const int TPB = 256;
    auto nb = [&](int n) { return (n + TPB - 1) / TPB; };

    // ---- prep: fragments, degrees, rf, h0 ----
    k_pack_frag<<<nb(128 * 8 * 512), TPB, 0, stream>>>(en_W2, w2frag, 64, 4, 128 * 8 * 512);
    k_pack_frag<<<nb(8 * 512),       TPB, 0, stream>>>(en_b2, b2frag, 64, 4, 8 * 512);
    k_pack_frag<<<nb(8 * 512),       TPB, 0, stream>>>(root_W, rootfrag, 64, 4, 8 * 512);
    k_pack_frag<<<nb(24 * 512),      TPB, 0, stream>>>(g_wih, wihfrag, 192, 12, 24 * 512);
    k_pack_frag<<<nb(24 * 512),      TPB, 0, stream>>>(g_whh, whhfrag, 192, 12, 24 * 512);

    k_zero_u<<<nb(N), TPB, 0, stream>>>((unsigned*)deg, N);
    k_deg<<<nb(E), TPB, 0, stream>>>(etgt, deg, E);
    k_invdeg<<<nb(N), TPB, 0, stream>>>(deg, invdeg, N);
    k_edge_rf<<<nb(E * 128), TPB, 0, stream>>>((const float*)d_in[2], en_W1, en_b1, rf, E);
    k_init_h<<<nb(N * 64), TPB, 0, stream>>>(x, hA, N);

    // ---- 3 message-passing layers ----
    float* hin  = hA;
    float* hout = hB;
    const int etiles = (E + 15) / 16;
    const int ntiles = (N + 15) / 16;
    for (int layer = 0; layer < 3; ++layer) {
        k_zero_f<<<nb(N * 64), TPB, 0, stream>>>(agg, N * 64);
        k_edge_msg<<<etiles, 32, 0, stream>>>(rf, hin, w2frag, b2frag, esrc, etgt, agg, E);
        float* outn = (layer == 2) ? (float*)d_out : nullptr;
        k_node_gru<<<ntiles, 32, 0, stream>>>(hin, agg, invdeg, rootfrag, wihfrag, whhfrag,
                                              conv_b, g_bih, g_bhh, hout, outn, N);
        float* t = hin; hin = hout; hout = t;
    }
    const float* nodes = hin;   // final node_reps

    // ---- Set2Set (3 steps) ----
    k_zero_f<<<nb(Bg * 128), TPB, 0, stream>>>(qs0, Bg * 128);
    k_zero_f<<<nb(Bg * 64),  TPB, 0, stream>>>(hh0, Bg * 64);
    k_zero_f<<<nb(Bg * 64),  TPB, 0, stream>>>(ccb, Bg * 64);
    float* qin = qs0; float* qout = qs1;
    float* hhin = hh0; float* hhout = hh1;
    for (int step = 0; step < 3; ++step) {
        k_lstm<<<nb(Bg * 64), TPB, 0, stream>>>(qin, l_wih, l_whh, l_bih, l_bhh,
                                                hhin, hhout, ccb, qout, Bg);
        k_s2s_init<<<nb(Bg * 64), TPB, 0, stream>>>(mx, denom, qout, Bg);
        k_dotmax<<<nb(N), TPB, 0, stream>>>(nodes, hhout, batch, evals, mx, N);
        k_expsum<<<nb(N), TPB, 0, stream>>>(evals, mx, batch, avals, denom, N);
        k_weighted<<<nb(N), TPB, 0, stream>>>(avals, denom, batch, nodes, qout, N);
        float* t = qin; qin = qout; qout = t;
        t = hhin; hhin = hhout; hhout = t;
    }

    // ---- readout MLP -> graph_rep at d_out + N*64 ----
    k_ro1<<<nb(Bg * 64), TPB, 0, stream>>>(qin, ro_W1, ro_b1, tbuf, Bg);
    k_ro2<<<nb(Bg * 64), TPB, 0, stream>>>(tbuf, ro_W2, ro_b2, (float*)d_out + (size_t)N * 64, Bg);
}
